// MLP_65901978190420
// MI455X (gfx1250) — matrix operand order repaired and checked by reading, not compiled, not run on hardware
//
#include <hip/hip_runtime.h>
#include <hip/hip_bf16.h>
#include <math.h>

// ---------------------------------------------------------------------------
// ZipNeRF pipeline for MI455X (gfx1250, wave32):
//   kernel 1: hash-grid trilinear encode + erf weighting -> f16 features
//   kernel 2: weight pre-pack (f32 [K][N] -> f16 transposed padded [Np][Kp])
//   kernel 3: fully fused 5-layer MLP on v_wmma_f32_16x16x32_f16,
//             all fragment loads are unconditional 16-byte vector loads.
// ---------------------------------------------------------------------------

typedef __attribute__((ext_vector_type(16))) _Float16 v16h;
typedef __attribute__((ext_vector_type(8)))  _Float16 v8h;
typedef __attribute__((ext_vector_type(4)))  _Float16 v4h;
typedef __attribute__((ext_vector_type(8)))  float    v8f;

#define T_HASH   2097152
#define NROWS    32768            // B*S = 1024*32
#define NSAMP    6
#define NLEVELS  10
#define FEAT_LD  64               // 40 features padded to 64 halfs per row

// ---------------- WMMA helpers (layouts per CDNA5 ISA) -----------

__device__ __forceinline__ v8f wmma32(v16h a, v16h b, v8f c) {
  return __builtin_amdgcn_wmma_f32_16x16x32_f16(
      false, a, false, b, (short)0, c, false, false);
}

// A-matrix 16x32 f16: lane L holds row M = L&15.
// lanes 0-15 : elems 0-7 -> K=k0+0..7,  elems 8-15 -> K=k0+16..23
// lanes 16-31: elems 0-7 -> K=k0+8..15, elems 8-15 -> K=k0+24..31
// => two contiguous 8-half (16B) vector loads per fragment.
__device__ __forceinline__ v16h load_frag_a(const _Float16* A, int ld, int lane, int k0) {
  const int row = lane & 15;
  const _Float16* p = A + row * ld + k0 + ((lane & 16) >> 1);
  const v8h lo = *(const v8h*)(p);
  const v8h hi = *(const v8h*)(p + 16);
  return __builtin_shufflevector(lo, hi, 0, 1, 2, 3, 4, 5, 6, 7,
                                 8, 9, 10, 11, 12, 13, 14, 15);
}

// B-matrix 32x16 f16 from pre-transposed padded weights Wt[Np][Kp]:
// lane L holds col N = L&15; same K order as A: lanes 0-15 K=k0+0..7,16..23,
// lanes 16-31 K=k0+8..15,24..31 => two 16B vector loads.
__device__ __forceinline__ v16h load_frag_b(const _Float16* Wt, int ldk, int lane,
                                            int n0, int k0) {
  const int col = n0 + (lane & 15);
  const _Float16* p = Wt + col * ldk + k0 + ((lane & 16) >> 1);
  const v8h lo = *(const v8h*)(p);
  const v8h hi = *(const v8h*)(p + 16);
  return __builtin_shufflevector(lo, hi, 0, 1, 2, 3, 4, 5, 6, 7,
                                 8, 9, 10, 11, 12, 13, 14, 15);
}

// ---------------- Kernel 1: hash-grid encode -> f16 features ----------------
__global__ void zipnerf_hash_kernel(const float* __restrict__ means,
                                    const float* __restrict__ stds,
                                    const float* __restrict__ emb,
                                    _Float16* __restrict__ feat_h) {
  const int t = blockIdx.x * blockDim.x + threadIdx.x;
  if (t >= NROWS * NLEVELS) return;
  const int row   = t / NLEVELS;
  const int level = t - row * NLEVELS;

  const float g = ldexpf(16.0f, level);          // RES[level] = 16 * 2^level
  const float4* __restrict__ E = (const float4*)emb + (size_t)level * T_HASH;

  float ax = 0.f, ay = 0.f, az = 0.f, aw = 0.f;
  const float sqrt8 = 2.8284271247461903f;

#pragma unroll
  for (int n = 0; n < NSAMP; ++n) {
    const int mi = (row * NSAMP + n) * 3;
    const float sd = stds[row * NSAMP + n];
    const float posx = (means[mi + 0] + 1.0f) * 0.5f * g - 0.5f;
    const float posy = (means[mi + 1] + 1.0f) * 0.5f * g - 0.5f;
    const float posz = (means[mi + 2] + 1.0f) * 0.5f * g - 0.5f;
    const float fx0 = floorf(posx), fy0 = floorf(posy), fz0 = floorf(posz);
    const float frx = posx - fx0, fry = posy - fy0, frz = posz - fz0;
    const int ix = (int)fx0, iy = (int)fy0, iz = (int)fz0;

    const float denom = fmaxf(sqrt8 * sd * g, 1e-10f);
    const float werf  = erff(1.0f / denom);

#pragma unroll
    for (int c8 = 0; c8 < 8; ++c8) {
      const int cx = (c8 >> 2) & 1, cy = (c8 >> 1) & 1, cz = c8 & 1;
      const unsigned ux = (unsigned)(ix + cx);
      const unsigned uy = (unsigned)(iy + cy);
      const unsigned uz = (unsigned)(iz + cz);
      const unsigned h  = ux ^ (uy * 2654435761u) ^ (uz * 805459861u);
      const int idx = (int)(h & (unsigned)(T_HASH - 1));
      const float w = (cx ? frx : 1.0f - frx) *
                      (cy ? fry : 1.0f - fry) *
                      (cz ? frz : 1.0f - frz) * werf;
      const float4 e = E[idx];               // global_load_b128
      ax += w * e.x; ay += w * e.y; az += w * e.z; aw += w * e.w;
    }
  }
  const float inv6 = 1.0f / 6.0f;
  v4h o;
  o[0] = (_Float16)(ax * inv6); o[1] = (_Float16)(ay * inv6);
  o[2] = (_Float16)(az * inv6); o[3] = (_Float16)(aw * inv6);
  *(v4h*)(feat_h + (size_t)row * FEAT_LD + level * 4) = o;

  if (level == 0) {                          // zero the K padding cols 40..63
    const v8h z = {};
    _Float16* pad = feat_h + (size_t)row * FEAT_LD + 40;
    *(v8h*)(pad) = z; *(v8h*)(pad + 8) = z; *(v8h*)(pad + 16) = z;
  }
}

// ---------------- Kernel 2: weight pre-pack ---------------------------------
// src: f32 row-major [K][N];  dst: f16 transposed padded [Np][Kp]
__global__ void pack_weights_kernel(const float* __restrict__ src,
                                    _Float16* __restrict__ dst,
                                    int K, int N, int Kp, int Np) {
  const int t = blockIdx.x * blockDim.x + threadIdx.x;
  if (t >= Np * Kp) return;
  const int n = t / Kp;
  const int k = t - n * Kp;
  float v = 0.0f;
  if (k < K && n < N) v = src[k * N + n];
  dst[t] = (_Float16)v;
}

// ---------------- Kernel 3: fused MLP (WMMA) --------------------------------
#define WPB 2
__global__ void __launch_bounds__(WPB * 32)
zipnerf_mlp_kernel(const _Float16* __restrict__ feat_h,
                   const float* __restrict__ viewdirs,
                   const _Float16* __restrict__ Wd0t, const float* __restrict__ bd0,
                   const _Float16* __restrict__ Wd1t, const float* __restrict__ bd1,
                   const _Float16* __restrict__ Wv0t, const float* __restrict__ bv0,
                   const _Float16* __restrict__ Wv1t, const float* __restrict__ bv1,
                   const _Float16* __restrict__ Wrt,  const float* __restrict__ br,
                   float* __restrict__ out) {
  __shared__ alignas(16) _Float16 s_h0 [WPB][16 * 64];    // layer1 out
  __shared__ alignas(16) _Float16 s_xin[WPB][16 * 288];   // x(256)+dir_enc(27)+pad
  __shared__ alignas(16) _Float16 s_h  [WPB][16 * 256];   // Wv0 out
  __shared__ alignas(16) _Float16 s_h2 [WPB][16 * 256];   // Wv1 out

  const int lane = threadIdx.x & 31;
  const int wv   = threadIdx.x >> 5;
  const int tile = blockIdx.x * WPB + wv;
  const int row0 = tile << 4;                 // rows [row0, row0+16)

  _Float16* h0  = &s_h0 [wv][0];
  _Float16* xin = &s_xin[wv][0];
  _Float16* h   = &s_h  [wv][0];
  _Float16* h2  = &s_h2 [wv][0];

  __builtin_prefetch(Wv0t, 0, 1);             // global_prefetch_b8
  __builtin_prefetch(Wv1t, 0, 1);

  const int mbase = (lane & 16) >> 1;         // C/D frag: m = v + 8*(lane>=16)

  // ---- Layer 1: feat[16,40p64] @ Wd0[40p64,64] + bd0, relu -> h0 -----------
#pragma unroll
  for (int nt = 0; nt < 4; ++nt) {
    v8f c = {};
#pragma unroll
    for (int kk = 0; kk < 64; kk += 32) {
      v16h a = load_frag_a(feat_h + (size_t)row0 * FEAT_LD, FEAT_LD, lane, kk);
      v16h b = load_frag_b(Wd0t, 64, lane, nt * 16, kk);
      c = wmma32(a, b, c);
    }
    const int n = nt * 16 + (lane & 15);
    const float bias = bd0[n];
#pragma unroll
    for (int v = 0; v < 8; ++v)
      h0[(v + mbase) * 64 + n] = (_Float16)fmaxf(c[v] + bias, 0.0f);
  }
  __syncthreads();

  // ---- Layer 2: h0[16,64] @ Wd1[64,256] + bd1 -> x (no relu), density ------
  for (int nt = 0; nt < 16; ++nt) {
    v8f c = {};
#pragma unroll
    for (int kk = 0; kk < 64; kk += 32) {
      v16h a = load_frag_a(h0, 64, lane, kk);
      v16h b = load_frag_b(Wd1t, 64, lane, nt * 16, kk);
      c = wmma32(a, b, c);
    }
    const int n = nt * 16 + (lane & 15);
    const float bias = bd1[n];
#pragma unroll
    for (int v = 0; v < 8; ++v) {
      const int m = v + mbase;
      const float x = c[v] + bias;
      xin[m * 288 + n] = (_Float16)x;
      if (nt == 0 && (lane & 15) == 0) {      // column 0 -> density
        const float z  = x - 1.0f;            // DENSITY_BIAS
        const float sp = (z > 20.0f) ? z : log1pf(expf(z));
        out[row0 + m] = sp;                   // softplus
      }
    }
  }

  // ---- dir_enc: 27 values per ray (constant across the 16-row tile) --------
  {
    const int bidx = row0 >> 5;               // 32 samples per ray
    const float d0 = viewdirs[bidx * 3 + 0];
    const float d1 = viewdirs[bidx * 3 + 1];
    const float d2 = viewdirs[bidx * 3 + 2];
    const int j = lane;                       // columns 256 + j, j = 0..31
    float val = 0.0f;
    if (j < 3) {
      val = (j == 0) ? d0 : ((j == 1) ? d1 : d2);
    } else if (j < 27) {
      const bool is_cos = (j >= 15);          // sin(xb + pi/2) == cos(xb)
      const int  ff  = is_cos ? (j - 15) : (j - 3);
      const int  deg = ff / 3, dim = ff % 3;
      const float dd = (dim == 0) ? d0 : ((dim == 1) ? d1 : d2);
      const float xb = dd * exp2f((float)deg);
      val = is_cos ? cosf(xb) : sinf(xb);
    }                                         // j >= 27 -> zero padding
#pragma unroll
    for (int r = 0; r < 16; ++r) xin[r * 288 + 256 + j] = (_Float16)val;
  }
  __syncthreads();

  // ---- Layer 3: x_in[16,283p288] @ Wv0[283p288,256] + bv0, relu -> h -------
  for (int nt = 0; nt < 16; ++nt) {
    v8f c = {};
#pragma unroll
    for (int kk = 0; kk < 288; kk += 32) {
      v16h a = load_frag_a(xin, 288, lane, kk);
      v16h b = load_frag_b(Wv0t, 288, lane, nt * 16, kk);
      c = wmma32(a, b, c);
    }
    const int n = nt * 16 + (lane & 15);
    const float bias = bv0[n];
#pragma unroll
    for (int v = 0; v < 8; ++v)
      h[(v + mbase) * 256 + n] = (_Float16)fmaxf(c[v] + bias, 0.0f);
  }
  __syncthreads();

  // ---- Layer 4 (skip concat): [h | x_in][16,539p544] @ Wv1 + bv1, relu -----
  for (int nt = 0; nt < 16; ++nt) {
    v8f c = {};
#pragma unroll
    for (int kk = 0; kk < 256; kk += 32) {           // h part: Wv1 rows 0..255
      v16h a = load_frag_a(h, 256, lane, kk);
      v16h b = load_frag_b(Wv1t, 544, lane, nt * 16, kk);
      c = wmma32(a, b, c);
    }
#pragma unroll
    for (int kk = 0; kk < 288; kk += 32) {           // x_in part: rows 256..538
      v16h a = load_frag_a(xin, 288, lane, kk);
      v16h b = load_frag_b(Wv1t, 544, lane, nt * 16, 256 + kk);
      c = wmma32(a, b, c);
    }
    const int n = nt * 16 + (lane & 15);
    const float bias = bv1[n];
#pragma unroll
    for (int v = 0; v < 8; ++v)
      h2[(v + mbase) * 256 + n] = (_Float16)fmaxf(c[v] + bias, 0.0f);
  }
  __syncthreads();

  // ---- Layer 5: h2[16,256] @ Wr[256,3p16] + br, sigmoid, rgb padding -------
  {
    v8f c = {};
#pragma unroll
    for (int kk = 0; kk < 256; kk += 32) {
      v16h a = load_frag_a(h2, 256, lane, kk);
      v16h b = load_frag_b(Wrt, 256, lane, 0, kk);
      c = wmma32(a, b, c);
    }
    const int n = lane & 15;
    if (n < 3) {
      const float bias = br[n];
#pragma unroll
      for (int v = 0; v < 8; ++v) {
        const int m = v + mbase;
        const float s = 1.0f / (1.0f + expf(-(c[v] + bias)));
        out[NROWS + (row0 + m) * 3 + n] = s * 1.002f - 0.001f;
      }
    }
  }
}

// ---------------- host launch ------------------------------------------------
extern "C" void kernel_launch(void* const* d_in, const int* in_sizes, int n_in,
                              void* d_out, int out_size, void* d_ws, size_t ws_size,
                              hipStream_t stream) {
  (void)in_sizes; (void)n_in; (void)out_size; (void)ws_size;
  // input order: rand, means, stds, viewdirs, embeddings,
  //              Wd0, bd0, Wd1, bd1, Wv0, bv0, Wv1, bv1, Wr, br
  const float* means    = (const float*)d_in[1];
  const float* stds     = (const float*)d_in[2];
  const float* viewdirs = (const float*)d_in[3];
  const float* emb      = (const float*)d_in[4];
  const float* Wd0 = (const float*)d_in[5];  const float* bd0 = (const float*)d_in[6];
  const float* Wd1 = (const float*)d_in[7];  const float* bd1 = (const float*)d_in[8];
  const float* Wv0 = (const float*)d_in[9];  const float* bv0 = (const float*)d_in[10];
  const float* Wv1 = (const float*)d_in[11]; const float* bv1 = (const float*)d_in[12];
  const float* Wr  = (const float*)d_in[13]; const float* br  = (const float*)d_in[14];
  float* out = (float*)d_out;

  // workspace layout (f16), all 16B-aligned
  char* ws = (char*)d_ws;
  _Float16* feat_h = (_Float16*)(ws);                         // 32768*64 = 4 MiB
  _Float16* Wd0t   = (_Float16*)(ws + 4194304);               //  64 x  64
  _Float16* Wd1t   = (_Float16*)(ws + 4194304 + 8192);        // 256 x  64
  _Float16* Wv0t   = (_Float16*)(ws + 4194304 + 40960);       // 256 x 288
  _Float16* Wv1t   = (_Float16*)(ws + 4194304 + 188416);      // 256 x 544
  _Float16* Wrt    = (_Float16*)(ws + 4194304 + 466944);      //  16 x 256

  // kernel 1: hash-grid encode
  const int total1 = NROWS * NLEVELS;
  zipnerf_hash_kernel<<<(total1 + 255) / 256, 256, 0, stream>>>(means, stds, emb, feat_h);

  // kernel 2: pack all five weight matrices (K,N) -> f16 [Np][Kp]
  auto pack = [&](const float* src, _Float16* dst, int K, int N, int Kp, int Np) {
    const int tot = Np * Kp;
    pack_weights_kernel<<<(tot + 255) / 256, 256, 0, stream>>>(src, dst, K, N, Kp, Np);
  };
  pack(Wd0,  Wd0t,  40,  64,  64,  64);
  pack(Wd1,  Wd1t,  64, 256,  64, 256);
  pack(Wv0,  Wv0t, 283, 256, 288, 256);
  pack(Wv1,  Wv1t, 539, 256, 544, 256);
  pack(Wr,   Wrt,  256,   3, 256,  16);

  // kernel 3: 2048 tiles of 16 rows, 2 waves (tiles) per 64-thread block
  zipnerf_mlp_kernel<<<(NROWS / 16) / WPB, WPB * 32, 0, stream>>>(
      feat_h, viewdirs, Wd0t, bd0, Wd1t, bd1, Wv0t, bv0, Wv1t, bv1, Wrt, br, out);
}
